// EasyControlNetwork_55104430407703
// MI455X (gfx1250) — matrix operand order repaired and checked by reading, not compiled, not run on hardware
//
#include <hip/hip_runtime.h>
#include <hip/hip_bf16.h>

typedef __attribute__((ext_vector_type(16))) _Float16 v16h;
typedef __attribute__((ext_vector_type(8)))  _Float16 v8h;
typedef __attribute__((ext_vector_type(8)))  float    v8f;
typedef __attribute__((ext_vector_type(4)))  float    v4f;
typedef __attribute__((ext_vector_type(4)))  unsigned int v4u;
typedef __attribute__((ext_vector_type(8)))  int      v8i;
typedef __attribute__((ext_vector_type(4)))  int      v4i;

#define B_   2
#define S_Tq 2048
#define S_Cc 1024
#define H_   16
#define D_   128
#define S_K  (S_Tq + S_Cc)
#define KT   32
#define QT   128
#define NW   8
#define NT   (S_K / KT)

// ---- CDNA5 data-mover feature detection (compile-safe fallbacks) ----
#if defined(__has_builtin)
#  if __has_builtin(__builtin_amdgcn_global_load_async_to_lds_b128)
#    define HAVE_ASYNC_COPY 1
#  else
#    define HAVE_ASYNC_COPY 0
#  endif
#  if __has_builtin(__builtin_amdgcn_tensor_load_to_lds)
#    define HAVE_TDM 1
#  else
#    define HAVE_TDM 0
#  endif
#  if __has_builtin(__builtin_amdgcn_s_wait_asynccnt)
#    define WAIT_ASYNC(n) __builtin_amdgcn_s_wait_asynccnt(n)
#  else
#    define WAIT_ASYNC(n) asm volatile("s_wait_asynccnt " #n ::: "memory")
#  endif
#  if __has_builtin(__builtin_amdgcn_s_wait_tensorcnt)
#    define WAIT_TENSOR(n) __builtin_amdgcn_s_wait_tensorcnt(n)
#  else
#    define WAIT_TENSOR(n) asm volatile("s_wait_tensorcnt " #n ::: "memory")
#  endif
#else
#  define HAVE_ASYNC_COPY 0
#  define HAVE_TDM 0
#  define WAIT_ASYNC(n)
#  define WAIT_TENSOR(n)
#endif

// typed address-space pointers for the async copy builtin
#define AS1V4I(p) ((__attribute__((address_space(1))) v4i*)(unsigned long long)(size_t)(p))
#define AS3V4I(p) ((__attribute__((address_space(3))) v4i*)(unsigned)(size_t)(p))

// =====================================================================
// Prepass: K_ext -> f16 [B,H,S_K,D] (contiguous key tiles per (b,h)),
//          V_ext -> f16 [B,H,D,S_K] (pre-transposed for PV B-fragments)
// =====================================================================
__global__ __launch_bounds__(256) void cvt_kv_kernel(
    const float* __restrict__ tk, const float* __restrict__ tv,
    const float* __restrict__ ck, const float* __restrict__ cv,
    _Float16* __restrict__ wsK, _Float16* __restrict__ wsVT)
{
  __shared__ alignas(32) _Float16 vt[KT][D_];
  const int tid = threadIdx.x;
  const int stile = blockIdx.x;
  const int hh = blockIdx.y, bb = blockIdx.z;
  const int bh = bb * H_ + hh;

  const int sk = tid >> 3;          // key row 0..31
  const int sd = (tid & 7) << 4;    // d offset 0..112
  const int s  = stile * KT + sk;

  const float *kr, *vr;
  if (s < S_Tq) {
    const long off = (((long)bb * S_Tq + s) * H_ + hh) * D_ + sd;
    kr = tk + off; vr = tv + off;
  } else {
    const long off = (((long)bb * S_Cc + (s - S_Tq)) * H_ + hh) * D_ + sd;
    kr = ck + off; vr = cv + off;
  }

  v8h kp0, kp1;
#pragma unroll
  for (int j = 0; j < 16; j += 4) {
    v4f kf = *(const v4f*)(kr + j);
    v4f vf = *(const v4f*)(vr + j);
#pragma unroll
    for (int e = 0; e < 4; ++e) {
      _Float16 kh16 = (_Float16)kf[e];
      if (j < 8) kp0[j + e] = kh16; else kp1[j - 8 + e] = kh16;
      vt[sk][sd + j + e] = (_Float16)vf[e];
    }
  }
  _Float16* ko = wsK + ((long)bh * S_K + s) * D_ + sd;
  *(v8h*)ko = kp0;
  *(v8h*)(ko + 8) = kp1;

  __syncthreads();

  const int dr = tid >> 1;
  const int kh = (tid & 1) << 4;
  v8h o0, o1;
#pragma unroll
  for (int e = 0; e < 8; ++e) {
    o0[e] = vt[kh + e][dr];
    o1[e] = vt[kh + 8 + e][dr];
  }
  _Float16* vo = wsVT + ((long)bh * D_ + dr) * S_K + stile * KT + kh;
  *(v8h*)vo = o0;
  *(v8h*)(vo + 8) = o1;
}

// =====================================================================
// Main flash-attention kernel: f16 WMMA + double-buffered TDM/async staging
// =====================================================================
__global__ __launch_bounds__(256) void fa_ext_attn_kernel(
    const float* __restrict__ tq,
    const _Float16* __restrict__ wsK,
    const _Float16* __restrict__ wsVT,
    const float* __restrict__ bp,
    float* __restrict__ out)
{
  __shared__ alignas(32) _Float16 ldsK[2][KT * D_];    // [buf][key][d]
  __shared__ alignas(32) _Float16 ldsVT[2][D_ * KT];   // [buf][d][key]
  __shared__ alignas(32) _Float16 ldsP[NW][16 * KT];   // per-wave P tile

  const int tid   = threadIdx.x;
  const int wave  = tid >> 5;
  const int lane  = tid & 31;
  const int lhalf = lane >> 4;
  const int lmod  = lane & 15;

  const int qtile = blockIdx.x;
  const int hh    = blockIdx.y;
  const int bb    = blockIdx.z;
  const int bh    = bb * H_ + hh;

  const float scale = 0.08838834764831845f;
  const float bval  = bp[0];

  const _Float16* kbase  = wsK  + (long)bh * S_K * D_;
  const _Float16* vtbase = wsVT + (long)bh * D_ * S_K;

  // stage key tile t into buffer t&1 (K via TDM, VT via async b128 copies)
  auto stage = [&](int t) {
    const int tile0 = t * KT;
    const int buf = t & 1;
#if HAVE_TDM
    if (wave == 0) {
      const unsigned ldsoff = (unsigned)(size_t)(&ldsK[buf][0]);
      const unsigned long long ga =
          (unsigned long long)(size_t)(kbase + (long)tile0 * D_);
      v4u g0;
      g0[0] = 1u;                                   // count=1, user mode
      g0[1] = ldsoff;                               // lds_addr
      g0[2] = (unsigned)ga;                         // global_addr[31:0]
      g0[3] = (unsigned)((ga >> 32) & 0x1FFFFFFu) | (2u << 30); // addr[56:32]|type=2
      v8i g1;
      g1[0] = (int)(1u << 16);                      // data_size=2B, no multicast
      g1[1] = (int)(128u << 16);                    // tensor_dim0 = 128
      g1[2] = (int)((unsigned)S_K << 16);           // tensor_dim1 = S_K
      g1[3] = (int)(128u << 16);                    // tile_dim0 = 128
      g1[4] = 32;                                   // tile_dim1 = 32
      g1[5] = 128;                                  // tensor_dim0_stride = 128
      g1[6] = 0;
      g1[7] = 0;
      v4i z4; z4[0] = 0; z4[1] = 0; z4[2] = 0; z4[3] = 0;
#if __clang_major__ >= 23
      v8i z8;
#pragma unroll
      for (int e = 0; e < 8; ++e) z8[e] = 0;
      __builtin_amdgcn_tensor_load_to_lds(g0, g1, z4, z4, z8, 0);
#else
      __builtin_amdgcn_tensor_load_to_lds(g0, g1, z4, z4, 0);
#endif
    }
#else
    {
      const v8h* g = (const v8h*)(kbase + (long)tile0 * D_) + tid * 2;
      v8h a = g[0], b = g[1];
      *(v8h*)&ldsK[buf][tid * 16]     = a;
      *(v8h*)&ldsK[buf][tid * 16 + 8] = b;
    }
#endif
    {
      const int dr = tid >> 1;
      const int kh = (tid & 1) << 4;
      const _Float16* g = vtbase + (long)dr * S_K + tile0 + kh;
      _Float16* l = &ldsVT[buf][dr * 32 + kh];
#if HAVE_ASYNC_COPY
      __builtin_amdgcn_global_load_async_to_lds_b128(AS1V4I(g), AS3V4I(l), 0, 0);
      __builtin_amdgcn_global_load_async_to_lds_b128(AS1V4I(g), AS3V4I(l), 16, 0);
#else
      v8h a = *(const v8h*)g;
      v8h b = *(const v8h*)(g + 8);
      *(v8h*)l = a;
      *(v8h*)(l + 8) = b;
#endif
    }
  };

  // ---------- load Q fragments (A layout: one contiguous 16-half K chunk per lane) ----------
  const int q0   = qtile * QT + wave * 16;
  const float* qptr = tq + (((long)bb * S_Tq + (q0 + lmod)) * H_ + hh) * D_;

  stage(0);   // prologue: tile 0 in flight while we set up Q

  v16h afrag[4];
#pragma unroll
  for (int c = 0; c < 4; ++c) {
    const int k0 = c * 32 + lhalf * 16;
    v4f f0 = *(const v4f*)(qptr + k0);
    v4f f1 = *(const v4f*)(qptr + k0 + 4);
    v4f f2 = *(const v4f*)(qptr + k0 + 8);
    v4f f3 = *(const v4f*)(qptr + k0 + 12);
#pragma unroll
    for (int e = 0; e < 4; ++e) {
      afrag[c][e]      = (_Float16)f0[e];
      afrag[c][4 + e]  = (_Float16)f1[e];
      afrag[c][8 + e]  = (_Float16)f2[e];
      afrag[c][12 + e] = (_Float16)f3[e];
    }
  }

  v8f acc[8];
  float m_i[8], l_i[8];
#pragma unroll
  for (int j = 0; j < 8; ++j) { m_i[j] = -3.402823466e+38f; l_i[j] = 0.0f; }
#pragma unroll
  for (int d = 0; d < 8; ++d)
#pragma unroll
    for (int j = 0; j < 8; ++j) acc[d][j] = 0.0f;

  for (int tile = 0; tile < NT; ++tile) {
    __syncthreads();   // all waves done reading the buffer stage(tile+1) overwrites

    if (tile + 1 < NT) {
      stage(tile + 1);                 // overlap next transfer with this compute
      WAIT_TENSOR(1);                  // tile's K in LDS; next TDM stays in flight
      WAIT_ASYNC(2);                   // tile's VT in LDS; next 2 copies in flight
    } else {
      WAIT_TENSOR(0);
      WAIT_ASYNC(0);
    }

    if (tile + 2 < NT) {               // prefetch t+2 into GL2
      __builtin_prefetch(kbase + (long)(tile + 2) * KT * D_ + tid * 16, 0, 0);
      __builtin_prefetch(vtbase + (long)(tid >> 1) * S_K + (tile + 2) * KT, 0, 0);
    }
    __syncthreads();   // staged data visible to all waves

    const _Float16* lk = ldsK[tile & 1];
    const _Float16* lv = ldsVT[tile & 1];

    // ---------- S = Q * K^T ----------
    v16h bk[8];
#pragma unroll
    for (int c = 0; c < 4; ++c) {
      bk[c]     = *(const v16h*)(&lk[lmod * D_ + c * 32 + lhalf * 16]);
      bk[4 + c] = *(const v16h*)(&lk[(16 + lmod) * D_ + c * 32 + lhalf * 16]);
    }
    v8f s0, s1;
#pragma unroll
    for (int j = 0; j < 8; ++j) { s0[j] = 0.0f; s1[j] = 0.0f; }
#pragma unroll
    for (int c = 0; c < 4; ++c) {
      s0 = __builtin_amdgcn_wmma_f32_16x16x32_f16(false, afrag[c], false, bk[c],
                                                  (short)0, s0, false, false);
      s1 = __builtin_amdgcn_wmma_f32_16x16x32_f16(false, afrag[c], false, bk[4 + c],
                                                  (short)0, s1, false, false);
    }

    const float bias = (tile * KT >= S_Tq) ? bval : 0.0f;

    // ---------- online softmax ----------
#pragma unroll
    for (int j = 0; j < 8; ++j) {
      float a0 = s0[j] * scale + bias;
      float a1 = s1[j] * scale + bias;
      float rmax = fmaxf(a0, a1);
#pragma unroll
      for (int msk = 8; msk >= 1; msk >>= 1)
        rmax = fmaxf(rmax, __shfl_xor(rmax, msk, 32));
      const float mnew  = fmaxf(m_i[j], rmax);
      const float alpha = __expf(m_i[j] - mnew);
      const float p0 = __expf(a0 - mnew);
      const float p1 = __expf(a1 - mnew);
      float rsum = p0 + p1;
#pragma unroll
      for (int msk = 8; msk >= 1; msk >>= 1)
        rsum += __shfl_xor(rsum, msk, 32);
      l_i[j] = l_i[j] * alpha + rsum;
      m_i[j] = mnew;
#pragma unroll
      for (int d = 0; d < 8; ++d) acc[d][j] *= alpha;
      const int prow = j + (lhalf << 3);
      ldsP[wave][prow * KT + lmod]      = (_Float16)p0;
      ldsP[wave][prow * KT + 16 + lmod] = (_Float16)p1;
    }

    // P tile is wave-private: wave-local DS fence instead of a block barrier
    asm volatile("s_wait_dscnt 0x0" ::: "memory");

    // ---------- P A-fragment ----------
    v16h pf;
    {
      v8h lo = *(const v8h*)(&ldsP[wave][lmod * KT + lhalf * 16]);
      v8h hi = *(const v8h*)(&ldsP[wave][lmod * KT + lhalf * 16 + 8]);
#pragma unroll
      for (int e = 0; e < 8; ++e) { pf[e] = lo[e]; pf[8 + e] = hi[e]; }
    }

    // ---------- O += P * V ----------
    v16h vfr[8];
#pragma unroll
    for (int d = 0; d < 8; ++d)
      vfr[d] = *(const v16h*)(&lv[(d * 16 + lmod) * KT + lhalf * 16]);
#pragma unroll
    for (int d = 0; d < 8; ++d)
      acc[d] = __builtin_amdgcn_wmma_f32_16x16x32_f16(false, pf, false, vfr[d],
                                                      (short)0, acc[d], false, false);
  }

  // ---------- epilogue ----------
  float rl[8];
#pragma unroll
  for (int j = 0; j < 8; ++j) rl[j] = 1.0f / l_i[j];
#pragma unroll
  for (int d = 0; d < 8; ++d)
#pragma unroll
    for (int j = 0; j < 8; ++j) {
      const int qq  = q0 + j + (lhalf << 3);
      const int col = hh * D_ + d * 16 + lmod;
      out[((long)bb * S_Tq + qq) * (H_ * D_) + col] = acc[d][j] * rl[j];
    }
}

extern "C" void kernel_launch(void* const* d_in, const int* in_sizes, int n_in,
                              void* d_out, int out_size, void* d_ws, size_t ws_size,
                              hipStream_t stream) {
  (void)in_sizes; (void)n_in; (void)out_size; (void)ws_size;
  const float* tq = (const float*)d_in[0];
  const float* tk = (const float*)d_in[1];
  const float* tv = (const float*)d_in[2];
  const float* ck = (const float*)d_in[3];
  const float* cv = (const float*)d_in[4];
  const float* bp = (const float*)d_in[5];
  float* out = (float*)d_out;

  _Float16* wsK  = (_Float16*)d_ws;
  _Float16* wsVT = wsK + (size_t)B_ * H_ * S_K * D_;

  dim3 pgrid(S_K / KT, H_, B_);    // (96, 16, 2)
  cvt_kv_kernel<<<pgrid, 256, 0, stream>>>(tk, tv, ck, cv, wsK, wsVT);

  dim3 grid(S_Tq / QT, H_, B_);    // (16, 16, 2)
  fa_ext_attn_kernel<<<grid, 256, 0, stream>>>(tq, wsK, wsVT, bp, out);
}
